// Conv2dQInt8ModuleBase_12043088298159
// MI455X (gfx1250) — 8-bit matrix operand order repaired and checked by reading, compiled, not run on hardware
//
#include <hip/hip_runtime.h>
#include <math.h>


typedef __attribute__((ext_vector_type(8))) int v8i;

#define NB    64
#define CIN   16
#define COUT  16
#define HIN   256
#define WIN   256
#define HOUT  254
#define WOUT  254
#define YPB   8
#define ROWS  (YPB + 2)
#define ROWB  (WIN * CIN)

__global__ __launch_bounds__(32)
void conv_q8_setup(const int* __restrict__ w, const float* __restrict__ bias,
                   int* __restrict__ wsA, float* __restrict__ wsC) {
    const int lane = threadIdx.x;
    const int n = lane & 15, half = lane >> 4;

    for (int kh = 0; kh < 3; ++kh) {
        for (int v = 0; v < 8; ++v) {
            int word = 0;
            if (v < 6) {
                const int kx  = v >> 1;
                const int ci0 = half * 8 + (v & 1) * 4;
                for (int j = 0; j < 4; ++j) {
                    int q = w[((n * CIN + (ci0 + j)) * 3 + kh) * 3 + kx] & 0xFF;
                    word |= q << (8 * j);
                }
            }
            wsA[(kh * 8 + v) * 32 + lane] = word;
        }
    }

    if (lane < COUT) {
        int sw = 0;
        for (int ci = 0; ci < CIN; ++ci)
            for (int kh = 0; kh < 3; ++kh)
                for (int kx = 0; kx < 3; ++kx)
                    sw += w[((lane * CIN + ci) * 3 + kh) * 3 + kx];
        float bq = rintf(bias[lane] * 10000.0f);
        bq = fminf(fmaxf(bq, -2147483648.0f), 2147483647.0f);
        wsC[lane] = 1e-4f * (-7.0f * (float)sw + 3024.0f) + bq * 1e-4f;
    }
}

__global__ __launch_bounds__(256)
void conv_q8_main(const int* __restrict__ in, const int* __restrict__ wsA,
                  const float* __restrict__ wsC, float* __restrict__ out) {
    __shared__ char smem[ROWS * ROWB + 128];

    const int tid = threadIdx.x;
    const int nb  = blockIdx.x;
    const int y0  = blockIdx.y * YPB;

    for (int i = 0; i < (ROWS * CIN * WIN / 4) / 256; ++i) {
        const int c   = tid + i * 256;
        const int x4  = (c & 63) * 4;
        const int ci  = (c >> 6) & 15;
        const int r   = c >> 10;
        const int yin = y0 + r;
        int4 v = make_int4(0, 0, 0, 0);
        if (yin < HIN)
            v = *(const int4*)(in + (((nb * CIN + ci) * HIN + yin) << 8) + x4);
        char* p = smem + r * ROWB + x4 * CIN + ci;
        p[0 * CIN] = (char)v.x;
        p[1 * CIN] = (char)v.y;
        p[2 * CIN] = (char)v.z;
        p[3 * CIN] = (char)v.w;
    }
    __syncthreads();

    const int wave = tid >> 5;
    const int lane = tid & 31;
    const int px   = lane & 15;
    const int half = lane >> 4;
    const int hmask = (half == 0) ? -1 : 0;

    v8i aW[3];
    for (int kh = 0; kh < 3; ++kh)
        for (int v = 0; v < 8; ++v)
            aW[kh][v] = wsA[(kh * 8 + v) * 32 + lane];
    const v8i aOnes = (v8i){0x01010101, 0x01010101, 0x01010101,
                            0x01010101, 0x01010101, 0x01010101, 0, 0};
    float cc[8];
    for (int r = 0; r < 8; ++r) cc[r] = wsC[r + 8 * half];

    for (int t = wave; t < YPB * 16; t += 8) {
        const int oy = t >> 4;
        const int x0 = (t & 15) << 4;

        v8i bf[3];
#pragma unroll
        for (int kh = 0; kh < 3; ++kh) {
            const int2* p = (const int2*)(smem + (oy + kh) * ROWB
                                               + (x0 + px) * CIN + half * 8);
            const int2 a0 = p[0];
            const int2 a1 = p[2];
            const int2 a2 = p[4];
            bf[kh][0] = a0.x;          bf[kh][1] = a0.y;
            bf[kh][2] = a1.x;          bf[kh][3] = a1.y;
            bf[kh][4] = a2.x;          bf[kh][5] = a2.y;
            bf[kh][6] = 0;             bf[kh][7] = 0;
        }

        v8i acc1 = {};
        v8i acc2 = {};
        acc1 = __builtin_amdgcn_wmma_i32_16x16x64_iu8(
                   true, aW[0], true, bf[0], acc1, false, false);
        acc2 = __builtin_amdgcn_wmma_i32_16x16x64_iu8(
                   true, aOnes, true, bf[0], acc2, false, false);
        acc1 = __builtin_amdgcn_wmma_i32_16x16x64_iu8(
                   true, aW[1], true, bf[1], acc1, false, false);
        acc2 = __builtin_amdgcn_wmma_i32_16x16x64_iu8(
                   true, aOnes, true, bf[1], acc2, false, false);
        acc1 = __builtin_amdgcn_wmma_i32_16x16x64_iu8(
                   true, aW[2], true, bf[2], acc1, false, false);
        acc2 = __builtin_amdgcn_wmma_i32_16x16x64_iu8(
                   true, aOnes, true, bf[2], acc2, false, false);

        const int x = x0 + px;
        const int y = y0 + oy;
        if (x < WOUT && y < HOUT) {
            for (int r = 0; r < 8; ++r) {
                const int cout = r + 8 * half;
                const float o = 1e-4f * (float)(acc1[r] - 3 * acc2[r]) + cc[r];
                out[((nb * COUT + cout) * HOUT + y) * WOUT + x] = o;
            }
        }
    }
}

extern "C" void kernel_launch(void* const* d_in, const int* in_sizes, int n_in,
                              void* d_out, int out_size, void* d_ws, size_t ws_size,
                              hipStream_t stream) {
    const int*   in   = (const int*)d_in[0];
    const int*   w    = (const int*)d_in[1];
    const float* bias = (const float*)d_in[2];

    int*   wsA = (int*)d_ws;
    float* wsC = (float*)((char*)d_ws + 3 * 8 * 32 * sizeof(int));

    conv_q8_setup<<<1, 32, 0, stream>>>(w, bias, wsA, wsC);

    dim3 grid(NB, (HOUT + YPB - 1) / YPB);
    conv_q8_main<<<grid, 256, 0, stream>>>(in, wsA, wsC, (float*)d_out);
}
